// Kernel3D_48962627175188
// MI455X (gfx1250) — matrix operand order repaired and checked by reading, not compiled, not run on hardware
//
#include <hip/hip_runtime.h>
#include <hip/hip_bf16.h>

typedef __attribute__((ext_vector_type(16))) __bf16 v16bf;
typedef __attribute__((ext_vector_type(8)))  float  v8f;

#define NG   4096
#define GRID 64
#define TDIM 8
#define ZT   512          // 64*8
#define KCH  64           // staged K per iteration (2 WMMA k-steps)
#define PA   72           // sA row pitch in bf16 elements (144B, 16B-aligned)
#define PB   72           // sB row pitch

__device__ __forceinline__ float gaussf(float mu, float s, float g) {
    // 1/sqrt(2*pi*s^2) * exp(-0.5*((mu-g)/s)^2)
    float inv = 0.3989422804014327f / s;
    float d   = (mu - g) / s;
    return inv * __expf(-0.5f * d * d);
}

// ---------------- Prologue 1: bxT[64][4096], byT[64][4096] (f32, n-contiguous)
__global__ void basis_xy_kernel(const float* __restrict__ mu,
                                const float* __restrict__ sigma,
                                float* __restrict__ bxT,
                                float* __restrict__ byT) {
    int i    = blockIdx.x * blockDim.x + threadIdx.x;   // 2*64*4096 threads
    int axis = i >> 18;                                 // 0 = x, 1 = y
    int rem  = i & 262143;
    int g    = rem >> 12;                               // grid coord 0..63
    int n    = rem & 4095;
    float v  = gaussf(mu[n * 3 + axis], sigma[n * 3 + axis], (float)g);
    float* dst = (axis == 0) ? bxT : byT;
    dst[g * 4096 + n] = v;
}

// ---------------- Prologue 2: BvT[zt][n] = bz[n,z]*x[n,t] (bf16, n-contiguous)
__global__ void bvt_kernel(const float* __restrict__ x,
                           const float* __restrict__ mu,
                           const float* __restrict__ sigma,
                           __bf16* __restrict__ BvT) {
    int i  = blockIdx.x * blockDim.x + threadIdx.x;     // 512*4096 threads
    int zt = i >> 12;
    int n  = i & 4095;
    int z  = zt >> 3, t = zt & 7;
    float v = gaussf(mu[n * 3 + 2], sigma[n * 3 + 2], (float)z) * x[n * TDIM + t];
    BvT[(size_t)zt * 4096 + n] = (__bf16)v;
}

// ---------------- Main: out[x, y, zt] = sum_n (bx*by)[n,x] * BvT[zt,n]
// grid = (4 zt-blocks of 128, 64 y), block = 256 (8 wave32)
__global__ __launch_bounds__(256)
void splat_wmma_kernel(const float* __restrict__ bxT,
                       const float* __restrict__ byT,
                       const __bf16* __restrict__ BvT,
                       float* __restrict__ out) {
    __shared__ __align__(16) __bf16 sA[64 * PA];   // [x][k]    9216 B
    __shared__ __align__(16) __bf16 sB[128 * PB];  // [zt][k]  18432 B

    const int tid    = threadIdx.x;
    const int lane   = tid & 31;
    const int w      = tid >> 5;
    const int y      = blockIdx.y;
    const int ztbase = blockIdx.x * 128;

    const int Mbase  = (w & 3) * 16;   // wave's x-tile
    const int Nstrip = (w >> 2) * 64;  // wave's 64-wide zt strip (4 N-tiles)

    v8f acc0 = 0.f, acc1 = 0.f, acc2 = 0.f, acc3 = 0.f;

    // staging roles (uniform across waves; no divergence around WMMA)
    const int ax  = tid >> 2;              // sA: x row 0..63
    const int aka = (tid & 3) * 16;        // sA: 16 k's per thread
    const int bc  = tid >> 1;              // sB: zt col 0..127
    const int bkb = (tid & 1) * 32;        // sB: 32 k's per thread

    const int mrow  = Mbase + (lane & 15);
    const int ahalf = (lane < 16) ? 0 : 8;
    const int bhalf = (lane < 16) ? 0 : 8;

    const float*  byrow = byT + (size_t)y * 4096;
    const float*  bxrow = bxT + (size_t)ax * 4096;
    const __bf16* bvrow = BvT + (size_t)(ztbase + bc) * 4096;

    for (int k0 = 0; k0 < NG; k0 += KCH) {
        // --- stage A: sA[x][k] = bf16(bx[n,x] * by[n,y]), n = k0+k
        {
            const float* bxp = bxrow + k0 + aka;
            const float* byp = byrow + k0 + aka;
            __bf16* ap = sA + ax * PA + aka;
#pragma unroll
            for (int j = 0; j < 16; ++j)
                ap[j] = (__bf16)(bxp[j] * byp[j]);
        }
        // --- stage B: copy BvT chunk (contiguous 16B vectors)
        {
            const uint4* src = (const uint4*)(bvrow + k0 + bkb);
            uint4* dst = (uint4*)(sB + bc * PB + bkb);
            dst[0] = src[0];
            dst[1] = src[1];
            dst[2] = src[2];
            dst[3] = src[3];
            if (k0 + KCH < NG)  // WGP-scope speculative prefetch of next chunk
                __builtin_prefetch(bvrow + k0 + KCH + bkb, 0, 3);
        }
        __syncthreads();

        // --- two K=32 sub-steps, 4 N-tiles each
        union { v16bf v; uint4 q[2]; } af;
        union { v16bf v; uint4 q[2]; } bf;
#define DO_TILE(KS, NT, ACC)                                                    \
        {                                                                       \
            const __bf16* bp =                                                  \
                sB + (Nstrip + (NT)*16 + (lane & 15)) * PB + (KS) + bhalf;      \
            bf.q[0] = *(const uint4*)(bp);                                      \
            bf.q[1] = *(const uint4*)(bp + 16);                                 \
            ACC = __builtin_amdgcn_wmma_f32_16x16x32_bf16(                      \
                false, af.v, false, bf.v, (short)0, ACC, false, false);         \
        }
#define DO_KSTEP(KS)                                                            \
        {                                                                       \
            const __bf16* apf = sA + mrow * PA + (KS) + ahalf;                  \
            af.q[0] = *(const uint4*)(apf);                                     \
            af.q[1] = *(const uint4*)(apf + 16);                                \
            DO_TILE(KS, 0, acc0)                                                \
            DO_TILE(KS, 1, acc1)                                                \
            DO_TILE(KS, 2, acc2)                                                \
            DO_TILE(KS, 3, acc3)                                                \
        }
        DO_KSTEP(0)
        DO_KSTEP(32)
#undef DO_KSTEP
#undef DO_TILE
        __syncthreads();
    }

    // --- epilogue: C layout vgpr r -> M = r + (lane<16?0:8), N = lane%16
    const int mlo  = Mbase + ((lane < 16) ? 0 : 8);
    const int cbas = y * ZT + ztbase + Nstrip + (lane & 15);
#pragma unroll
    for (int r = 0; r < 8; ++r) {
        const size_t row = (size_t)(mlo + r) * (GRID * ZT);
        out[row + cbas +  0] = acc0[r];
        out[row + cbas + 16] = acc1[r];
        out[row + cbas + 32] = acc2[r];
        out[row + cbas + 48] = acc3[r];
    }
}

extern "C" void kernel_launch(void* const* d_in, const int* in_sizes, int n_in,
                              void* d_out, int out_size, void* d_ws, size_t ws_size,
                              hipStream_t stream) {
    const float* x     = (const float*)d_in[0];   // (4096, 8)
    const float* mu    = (const float*)d_in[1];   // (4096, 3)
    const float* sigma = (const float*)d_in[2];   // (4096, 3)
    float* out = (float*)d_out;                   // (64,64,64,8)

    char* ws = (char*)d_ws;
    float*  bxT = (float*)(ws);                   // 1 MB: [64][4096]
    float*  byT = (float*)(ws + (1 << 20));       // 1 MB: [64][4096]
    __bf16* BvT = (__bf16*)(ws + (2 << 20));      // 4 MB: [512][4096]

    basis_xy_kernel<<<2048, 256, 0, stream>>>(mu, sigma, bxT, byT);
    bvt_kernel<<<8192, 256, 0, stream>>>(x, mu, sigma, BvT);
    splat_wmma_kernel<<<dim3(4, 64), 256, 0, stream>>>(bxT, byT, BvT, out);
}
